// RNN_75823352644143
// MI455X (gfx1250) — compile-verified, matrix operand order checked by reading, not run on hardware
//
#include <hip/hip_runtime.h>
#include <math.h>

// ---------------------------------------------------------------------------
// Seq2seq GRU (H=1024, L=2, S=T=512, V=32000) for MI455X / gfx1250, wave32.
//  - Batched input-gate GEMMs + vocab projection: v_wmma_f32_16x16x32_bf16
//    fed from LDS, staged with async global->LDS loads (ASYNCcnt) and
//    double-buffering.
//  - Sequential h-recurrence: bf16 weights (halved L2 traffic), persistent
//    kernel with device-wide barrier, s_sleep spin.
// ---------------------------------------------------------------------------

typedef unsigned short u16;
typedef unsigned int   u32;

typedef __attribute__((ext_vector_type(16))) __bf16 bf16x16;
typedef __attribute__((ext_vector_type(8)))  __bf16 bf16x8;
typedef __attribute__((ext_vector_type(8)))  float  f32x8;
typedef int v4i_vs __attribute__((vector_size(16)));   // matches builtin param

#define HDIM 1024
#define H3   3072
#define SEQ  512
#define VOC  32000
#define SCAN_BLOCKS 32
#define BK   32
#define LDSS 40          // LDS row stride in bf16: 80B, 16B-aligned, bank-spread

#ifndef __has_builtin
#define __has_builtin(x) 0
#endif
#if __has_builtin(__builtin_amdgcn_global_load_async_to_lds_b128) && \
    __has_builtin(__builtin_amdgcn_s_wait_asynccnt)
#define HAVE_ASYNC_LDS 1
#else
#define HAVE_ASYNC_LDS 0
#endif

__device__ __forceinline__ u16 f2bf(float f) {
  u32 u = __float_as_uint(f);
  u32 r = u + 0x7FFFu + ((u >> 16) & 1u);   // round-to-nearest-even
  return (u16)(r >> 16);
}
__device__ __forceinline__ float bf2f(u16 s) {
  return __uint_as_float(((u32)s) << 16);
}

// 16-byte chunk: global -> LDS (async on gfx1250, sync fallback otherwise)
__device__ __forceinline__ void stage_chunk(const u16* g, u16* l) {
#if HAVE_ASYNC_LDS
  __builtin_amdgcn_global_load_async_to_lds_b128(
      (__attribute__((address_space(1))) v4i_vs*)g,
      (__attribute__((address_space(3))) v4i_vs*)l, 0, 0);
#else
  struct alignas(16) B128S { u32 x, y, z, w; };
  *(B128S*)l = *(const B128S*)g;
#endif
}

__device__ __forceinline__ void wait_async(int n) {
#if HAVE_ASYNC_LDS
  if (n == 0)      __builtin_amdgcn_s_wait_asynccnt(0);
  else if (n == 1) __builtin_amdgcn_s_wait_asynccnt(1);
  else             __builtin_amdgcn_s_wait_asynccnt(2);
#else
  (void)n;
#endif
}

// --------------------------- fp32 -> bf16 convert ---------------------------
__global__ void cvt_bf16_kernel(const float* __restrict__ in,
                                u16* __restrict__ out, size_t n) {
  size_t i = (size_t)blockIdx.x * blockDim.x + threadIdx.x;
  if (i < n) out[i] = f2bf(in[i]);
}

// --------------------------- embedding gather ------------------------------
// dec_shift=1: row 0 uses BOS(=0), row t uses tokens[t-1] (teacher forcing)
__global__ void embed_kernel(const int* __restrict__ tokens,
                             const float* __restrict__ emb,
                             u16* __restrict__ out, int rows, int dec_shift) {
  int idx = blockIdx.x * blockDim.x + threadIdx.x;
  if (idx >= rows * HDIM) return;
  int row = idx >> 10, h = idx & (HDIM - 1);
  int tok = dec_shift ? ((row == 0) ? 0 : tokens[row - 1]) : tokens[row];
  out[idx] = f2bf(emb[(size_t)tok * HDIM + h]);
}

// ------------------ WMMA bf16 GEMM with bias, LDS-staged -------------------
// C[M,N] = A[M,K] * B[N,K]^T + bias[N];  A,B bf16 K-major, C f32.
// Block: 8 waves = 2(M) x 4(N) 16x16 tiles over a 32M x 64N block tile.
// Per BK=32 chunk: 2KB A + 4KB B staged to LDS by 384 per-lane async 16B
// loads (waves 0-3: 2 instrs, waves 4-7: 1), double-buffered against WMMA.
__global__ void __launch_bounds__(256)
wmma_gemm_bias(const u16* __restrict__ A, const u16* __restrict__ B,
               const float* __restrict__ bias, float* __restrict__ C,
               int K, int N) {
  __shared__ __align__(16) u16 Alds[2][32 * LDSS];
  __shared__ __align__(16) u16 Blds[2][64 * LDSS];

  const int tid  = threadIdx.x;
  const int lane = tid & 31;
  const int wave = tid >> 5;
  const int bm = blockIdx.x * 32;
  const int bn = blockIdx.y * 64;

  // staging chunk map: c in [0,384): c<128 -> A row c>>2 seg c&3,
  //                                  else  -> B row (c-128)>>2 seg (c-128)&3
  auto gsrc = [&](int c, int k0) -> const u16* {
    if (c < 128) { int r = c >> 2, s = c & 3;
                   return A + (size_t)(bm + r) * K + k0 + s * 8; }
    int cc = c - 128; int r = cc >> 2, s = cc & 3;
    return B + (size_t)(bn + r) * K + k0 + s * 8;
  };
  auto ldst = [&](int c, int buf) -> u16* {
    if (c < 128) { int r = c >> 2, s = c & 3; return &Alds[buf][r * LDSS + s * 8]; }
    int cc = c - 128; int r = cc >> 2, s = cc & 3;
    return &Blds[buf][r * LDSS + s * 8];
  };
  auto stage = [&](int buf, int k0) {
    stage_chunk(gsrc(tid, k0), ldst(tid, buf));
    if (tid < 128) stage_chunk(gsrc(tid + 256, k0), ldst(tid + 256, buf));
  };

  // per-wave fragment coordinates (A 16x32 layout; B mirrored over N)
  const int arow = (wave & 1) * 16 + (lane & 15);
  const int brow = (wave >> 1) * 16 + (lane & 15);
  const int kbe  = (lane >> 4) * 8;          // element offset of 8-elem group

  f32x8 acc = {};
  const int nk = K / BK;
  stage(0, 0);
  for (int kc = 0; kc < nk; ++kc) {
    const int buf = kc & 1;
    if (kc + 1 < nk) {
      stage(buf ^ 1, (kc + 1) * BK);
      wait_async(wave < 4 ? 2 : 1);          // only newly-issued remain
    } else {
      wait_async(0);
    }
    __syncthreads();

    const u16* ap = &Alds[buf][arow * LDSS + kbe];
    const u16* bp = &Blds[buf][brow * LDSS + kbe];
    bf16x8 al = *(const bf16x8*)(ap);
    bf16x8 ah = *(const bf16x8*)(ap + 16);
    bf16x8 bl = *(const bf16x8*)(bp);
    bf16x8 bh = *(const bf16x8*)(bp + 16);
    bf16x16 a = __builtin_shufflevector(al, ah, 0, 1, 2, 3, 4, 5, 6, 7,
                                        8, 9, 10, 11, 12, 13, 14, 15);
    bf16x16 b = __builtin_shufflevector(bl, bh, 0, 1, 2, 3, 4, 5, 6, 7,
                                        8, 9, 10, 11, 12, 13, 14, 15);
    acc = __builtin_amdgcn_wmma_f32_16x16x32_bf16(
        /*neg_a=*/false, a, /*neg_b=*/false, b,
        /*c_mod=*/(short)0, acc, /*reuse_a=*/false, /*reuse_b=*/false);

    __syncthreads();                         // protect buf before re-stage
  }

  const int m0 = bm + (wave & 1) * 16;
  const int n0 = bn + (wave >> 1) * 16;
  const float bv = bias[n0 + (lane & 15)];
  // C/D layout: VGPR i -> M = m0 + i + (lane>=16 ? 8 : 0), N = n0 + (lane&15)
  const int rbase = m0 + (lane >> 4) * 8;
  const int c = n0 + (lane & 15);
#pragma unroll
  for (int i = 0; i < 8; ++i)
    C[(size_t)(rbase + i) * N + c] = acc[i] + bv;
}

// ------------------------- sequential GRU recurrence -----------------------
// 32 blocks x 256 threads = 8192 lanes: output j = gtid>>3 (1024 outputs),
// 8 lanes per output each covering 128 K-elements of the bf16 Whh rows
// (r,z,n). Device-wide barrier between steps (monotonic counter).
__global__ void gru_scan(const u16* __restrict__ Whh,    // [3H,H] bf16
                         const float* __restrict__ bhh,  // [3H]
                         const float* __restrict__ gi,   // [SEQ,3H] precomputed
                         const float* __restrict__ h0,   // [H]
                         float* hbuf,                    // [2,H] double buffer
                         u16* __restrict__ out_bf,       // [SEQ,H] bf16
                         float* h_final,                 // [H] or nullptr
                         int* bar, int steps) {
  __shared__ float sr[256], sz[256], sn[256];
  const int tid = threadIdx.x;
  const int gtid = blockIdx.x * 256 + tid;
  const int j = gtid >> 3;
  const int part = gtid & 7;
  const int koff = part * 128;
  const u32* w0 = (const u32*)(Whh + ((size_t)j) * HDIM + koff);
  const u32* w1 = (const u32*)(Whh + ((size_t)(HDIM + j)) * HDIM + koff);
  const u32* w2 = (const u32*)(Whh + ((size_t)(2 * HDIM + j)) * HDIM + koff);

  for (int t = 0; t < steps; ++t) {
    const float* hp = (t == 0) ? h0 : (hbuf + ((t - 1) & 1) * HDIM);
    float s0 = 0.f, s1 = 0.f, s2 = 0.f;
#pragma unroll 8
    for (int k = 0; k < 64; ++k) {           // 128 bf16 = 64 dwords per row
      u32 a = w0[k], b = w1[k], c = w2[k];
      float hlo = hp[koff + 2 * k];
      float hhi = hp[koff + 2 * k + 1];
      s0 += bf2f((u16)a) * hlo + bf2f((u16)(a >> 16)) * hhi;
      s1 += bf2f((u16)b) * hlo + bf2f((u16)(b >> 16)) * hhi;
      s2 += bf2f((u16)c) * hlo + bf2f((u16)(c >> 16)) * hhi;
    }
    sr[tid] = s0; sz[tid] = s1; sn[tid] = s2;
    __syncthreads();
    if (part == 0) {
      for (int p = 1; p < 8; ++p) { s0 += sr[tid + p]; s1 += sz[tid + p]; s2 += sn[tid + p]; }
      const float* git = gi + (size_t)t * H3;
      float r = 1.0f / (1.0f + __expf(-(git[j] + s0 + bhh[j])));
      float z = 1.0f / (1.0f + __expf(-(git[HDIM + j] + s1 + bhh[HDIM + j])));
      float n = tanhf(git[2 * HDIM + j] + r * (s2 + bhh[2 * HDIM + j]));
      float hn = (1.0f - z) * n + z * hp[j];
      hbuf[(t & 1) * HDIM + j] = hn;
      out_bf[(size_t)t * HDIM + j] = f2bf(hn);
      if (h_final && t == steps - 1) h_final[j] = hn;
    }
    // ---- device-wide barrier (monotonic counter, no reset race) ----
    __threadfence();
    __syncthreads();
    if (tid == 0) {
      atomicAdd(bar, 1);
      volatile int* vb = bar;
      const int target = SCAN_BLOCKS * (t + 1);
      while (*vb < target) __builtin_amdgcn_s_sleep(1);
    }
    __syncthreads();
    __threadfence();
  }
}

// ----------------------------- row log-softmax -----------------------------
__global__ void log_softmax_rows(float* __restrict__ logits, int N) {
  const int row = blockIdx.x;
  float* p = logits + (size_t)row * N;
  __shared__ float red[256];
  float m = -INFINITY;
  for (int i = threadIdx.x; i < N; i += 256) m = fmaxf(m, p[i]);
  red[threadIdx.x] = m;
  __syncthreads();
  for (int s = 128; s; s >>= 1) {
    if (threadIdx.x < s) red[threadIdx.x] = fmaxf(red[threadIdx.x], red[threadIdx.x + s]);
    __syncthreads();
  }
  m = red[0];
  __syncthreads();
  float sum = 0.f;
  for (int i = threadIdx.x; i < N; i += 256) sum += __expf(p[i] - m);
  red[threadIdx.x] = sum;
  __syncthreads();
  for (int s = 128; s; s >>= 1) {
    if (threadIdx.x < s) red[threadIdx.x] += red[threadIdx.x + s];
    __syncthreads();
  }
  const float lse = m + __logf(red[0]);
  for (int i = threadIdx.x; i < N; i += 256) p[i] = p[i] - lse;
}

// ------------------------------- host driver -------------------------------
extern "C" void kernel_launch(void* const* d_in, const int* in_sizes, int n_in,
                              void* d_out, int out_size, void* d_ws, size_t ws_size,
                              hipStream_t stream) {
  const int*   src_tok  = (const int*)d_in[0];
  const int*   tgt_tok  = (const int*)d_in[1];
  const float* src_emb  = (const float*)d_in[2];
  const float* tgt_emb  = (const float*)d_in[3];
  const float* enc_Wih  = (const float*)d_in[4];
  const float* enc_Whh  = (const float*)d_in[5];
  const float* enc_bih  = (const float*)d_in[6];
  const float* enc_bhh  = (const float*)d_in[7];
  const float* dec_Wih  = (const float*)d_in[8];
  const float* dec_Whh  = (const float*)d_in[9];
  const float* dec_bih  = (const float*)d_in[10];
  const float* dec_bhh  = (const float*)d_in[11];
  const float* out_W    = (const float*)d_in[12];
  const float* out_b    = (const float*)d_in[13];
  float* out = (float*)d_out;

  char* ws = (char*)d_ws;
  size_t off = 0;
  auto walloc = [&](size_t bytes) -> void* {
    void* p = ws + off;
    off += (bytes + 255) & ~(size_t)255;
    return p;
  };
  const size_t gru_w_elems = 2ull * H3 * HDIM;          // per [L,3H,H] tensor
  u16*   w_enc_ih = (u16*)walloc(gru_w_elems * 2);
  u16*   w_enc_hh = (u16*)walloc(gru_w_elems * 2);
  u16*   w_dec_ih = (u16*)walloc(gru_w_elems * 2);
  u16*   w_dec_hh = (u16*)walloc(gru_w_elems * 2);
  u16*   w_out    = (u16*)walloc((size_t)VOC * HDIM * 2);
  u16*   xa       = (u16*)walloc((size_t)SEQ * HDIM * 2);
  u16*   xb       = (u16*)walloc((size_t)SEQ * HDIM * 2);
  float* gi       = (float*)walloc((size_t)SEQ * H3 * 4);
  float* hbuf     = (float*)walloc(2 * HDIM * 4);
  float* h0zero   = (float*)walloc(HDIM * 4);
  float* encfin   = (float*)walloc(2 * HDIM * 4);
  int*   bar      = (int*)walloc(256);
  (void)off; (void)ws_size; (void)in_sizes; (void)n_in; (void)out_size;

  // --- weight fp32 -> bf16 (deterministic, every call) ---
  {
    const size_t n = gru_w_elems;
    dim3 b(256), g((unsigned)((n + 255) / 256));
    cvt_bf16_kernel<<<g, b, 0, stream>>>(enc_Wih, w_enc_ih, n);
    cvt_bf16_kernel<<<g, b, 0, stream>>>(enc_Whh, w_enc_hh, n);
    cvt_bf16_kernel<<<g, b, 0, stream>>>(dec_Wih, w_dec_ih, n);
    cvt_bf16_kernel<<<g, b, 0, stream>>>(dec_Whh, w_dec_hh, n);
    const size_t no = (size_t)VOC * HDIM;
    cvt_bf16_kernel<<<dim3((unsigned)((no + 255) / 256)), b, 0, stream>>>(out_W, w_out, no);
  }
  (void)hipMemsetAsync(h0zero, 0, HDIM * sizeof(float), stream);

  const dim3 gemm_gi_grid(SEQ / 32, H3 / 64), gemm_block(256);
  const dim3 embed_grid((SEQ * HDIM + 255) / 256);

  // -------------------- encoder --------------------
  embed_kernel<<<embed_grid, 256, 0, stream>>>(src_tok, src_emb, xa, SEQ, 0);

  // enc layer 0: input xa -> output xb, final hidden -> encfin[0]
  wmma_gemm_bias<<<gemm_gi_grid, gemm_block, 0, stream>>>(xa, w_enc_ih, enc_bih, gi, HDIM, H3);
  (void)hipMemsetAsync(bar, 0, sizeof(int), stream);
  gru_scan<<<SCAN_BLOCKS, 256, 0, stream>>>(w_enc_hh, enc_bhh, gi, h0zero,
                                            hbuf, xb, encfin, bar, SEQ);
  // enc layer 1: input xb -> output xa, final hidden -> encfin[1]
  wmma_gemm_bias<<<gemm_gi_grid, gemm_block, 0, stream>>>(xb, w_enc_ih + gru_w_elems / 2,
                                                          enc_bih + H3, gi, HDIM, H3);
  (void)hipMemsetAsync(bar, 0, sizeof(int), stream);
  gru_scan<<<SCAN_BLOCKS, 256, 0, stream>>>(w_enc_hh + gru_w_elems / 2, enc_bhh + H3,
                                            gi, h0zero, hbuf, xa, encfin + HDIM, bar, SEQ);

  // -------------------- decoder (teacher forced) --------------------
  embed_kernel<<<embed_grid, 256, 0, stream>>>(tgt_tok, tgt_emb, xa, SEQ, 1);

  // dec layer 0: h0 = encfin[0], input xa -> output xb
  wmma_gemm_bias<<<gemm_gi_grid, gemm_block, 0, stream>>>(xa, w_dec_ih, dec_bih, gi, HDIM, H3);
  (void)hipMemsetAsync(bar, 0, sizeof(int), stream);
  gru_scan<<<SCAN_BLOCKS, 256, 0, stream>>>(w_dec_hh, dec_bhh, gi, encfin,
                                            hbuf, xb, nullptr, bar, SEQ);
  // dec layer 1: h0 = encfin[1], input xb -> output xa (= dec_top bf16)
  wmma_gemm_bias<<<gemm_gi_grid, gemm_block, 0, stream>>>(xb, w_dec_ih + gru_w_elems / 2,
                                                          dec_bih + H3, gi, HDIM, H3);
  (void)hipMemsetAsync(bar, 0, sizeof(int), stream);
  gru_scan<<<SCAN_BLOCKS, 256, 0, stream>>>(w_dec_hh + gru_w_elems / 2, dec_bhh + H3,
                                            gi, encfin + HDIM, hbuf, xa, nullptr, bar, SEQ);

  // -------------------- vocab projection + log-softmax --------------------
  wmma_gemm_bias<<<dim3(SEQ / 32, VOC / 64), gemm_block, 0, stream>>>(
      xa, w_out, out_b, out, HDIM, VOC);
  log_softmax_rows<<<SEQ, 256, 0, stream>>>(out, VOC);
}
